// ESBN_39161511805362
// MI455X (gfx1250) — matrix operand order repaired and checked by reading, not compiled, not run on hardware
//
#include <hip/hip_runtime.h>

typedef __bf16 bf16;
typedef __attribute__((ext_vector_type(16))) __bf16 v16bf;
typedef __attribute__((ext_vector_type(8)))  float  v8f;
typedef __attribute__((ext_vector_type(4)))  int    v4i;

#define B_SZ   512
#define T_SZ   32
#define D_SZ   1024
#define HID    512
#define Z_SZ   128
#define KEY    256
#define KRP    288     // key_r K padded to mult of 32 (257 -> 288)
#define HEADN  320     // 256 keyw + 1 gate + 4 y, padded to mult of 64
#define GATESN 2048
#define SLOTS  33
#define NEGF   (-1000000000.0f)

#ifndef __has_builtin
#define __has_builtin(x) 0
#endif

// Async global->LDS staging (gfx1250, ASYNCcnt-tracked). The builtin exists in
// this toolchain; its params are v4i pointers in AS1 (global) / AS3 (LDS).
#if __has_builtin(__builtin_amdgcn_global_load_async_to_lds_b128)
#define ESBN_ASYNC 1
#else
#define ESBN_ASYNC 0
#endif

#if __has_builtin(__builtin_amdgcn_s_wait_asynccnt)
#define ESBN_WAIT_ASYNC() __builtin_amdgcn_s_wait_asynccnt(0)
#else
#define ESBN_WAIT_ASYNC() asm volatile("s_wait_asynccnt 0x0" ::: "memory")
#endif

// ---------------------------------------------------------------------------
// bf16 WMMA GEMM: C = act(A(MxK) @ W(KxN) + bias), all operands bf16, fp32 acc.
// 256 threads = 8 wave32 waves; block tile 64x64; wave tile 16x32 (two 16x16
// accumulators sharing one A fragment); K step 32.
// Double-buffered LDS: tile k+1 streams in (async-to-LDS for A) while WMMAs
// consume tile k. Fragment element e -> K = 16*half + e for A and B.
// ---------------------------------------------------------------------------
template <int OUT_BF16, int RELU>
__global__ __launch_bounds__(256) void esbn_gemm(
    const bf16* __restrict__ A, const bf16* __restrict__ W,
    const float* __restrict__ bias, void* __restrict__ Cout,
    int M, int N, int K, int lda, int ldw, int ldc)
{
  __shared__ bf16 sA[2][64][32];     // [buf][m][k]
  __shared__ bf16 sBT[2][64][40];    // [buf][n][k], padded row

  const int tid  = threadIdx.x;
  const int wave = tid >> 5;
  const int lane = tid & 31;
  const int half = lane >> 4;
  const int l16  = lane & 15;

  const int m0 = blockIdx.x * 64;
  const int n0 = blockIdx.y * 64;
  const int wm = (wave >> 1) * 16;   // 0,16,32,48
  const int wn = (wave & 1) * 32;    // 0,32

  v8f acc0 = {};
  v8f acc1 = {};

  // A tile: 64x32, one 16B chunk per thread. B tile: 32x64, 8 elems per thread.
  const int ar = tid >> 2;
  const int ac = (tid & 3) * 8;
  const int br = tid >> 3;
  const int bc = (tid & 7) * 8;

  const bf16* aSrc = A + (size_t)(m0 + ar) * lda + ac;
  const bf16* bSrc = W + (size_t)br * ldw + n0 + bc;

  auto stage_tile = [&](int kk, int buf) {
#if ESBN_ASYNC
    __builtin_amdgcn_global_load_async_to_lds_b128(
        (__attribute__((address_space(1))) v4i*)(aSrc + kk),
        (__attribute__((address_space(3))) v4i*)&sA[buf][ar][ac],
        /*offset=*/0, /*cpol=*/0);
#else
    *(uint4*)&sA[buf][ar][ac] = *(const uint4*)(aSrc + kk);
#endif
    const bf16* s = bSrc + (size_t)kk * ldw;
    #pragma unroll
    for (int i = 0; i < 8; ++i) sBT[buf][bc + i][br] = s[i];   // transpose store
  };

  stage_tile(0, 0);
  int buf = 0;

  for (int kk = 0; kk < K; kk += 32) {
#if ESBN_ASYNC
    ESBN_WAIT_ASYNC();            // A tile for 'buf' fully landed in LDS
#endif
    __syncthreads();              // B tile stores + all waves synced

    if (kk + 32 < K) stage_tile(kk + 32, buf ^ 1);   // overlaps WMMA below

    // A fragment: lanes0-15 M=l16 K={0..15}; lanes16-31 K={16..31}
    union { v16bf v; unsigned u[8]; } fa;
    #pragma unroll
    for (int p = 0; p < 8; ++p) {
      const int kb = 2 * p + 16 * half;
      fa.u[p] = *(const unsigned*)&sA[buf][wm + l16][kb];
    }
    // B fragments: element e -> K = e + 16*half, N = l16
    union { v16bf v; unsigned u[8]; } fb0, fb1;
    #pragma unroll
    for (int p = 0; p < 8; ++p) {
      const int kb = 2 * p + 16 * half;
      fb0.u[p] = *(const unsigned*)&sBT[buf][wn + l16][kb];
      fb1.u[p] = *(const unsigned*)&sBT[buf][wn + 16 + l16][kb];
    }

    acc0 = __builtin_amdgcn_wmma_f32_16x16x32_bf16(
        false, fa.v, false, fb0.v, (short)0, acc0, false, false);
    acc1 = __builtin_amdgcn_wmma_f32_16x16x32_bf16(
        false, fa.v, false, fb1.v, (short)0, acc1, false, false);

    buf ^= 1;
  }

  // Epilogue. C/D layout: lane -> N=l16 (+16 for 2nd tile), VGPR r -> M=r+8*half
  const int nA = n0 + wn + l16;
  const int nB = n0 + wn + 16 + l16;
  const float bA = bias[nA];
  const float bB = bias[nB];
  #pragma unroll
  for (int r = 0; r < 8; ++r) {
    const int m = m0 + wm + r + 8 * half;
    float v0 = acc0[r] + bA;
    float v1 = acc1[r] + bB;
    if (RELU) { v0 = fmaxf(v0, 0.f); v1 = fmaxf(v1, 0.f); }
    if (OUT_BF16) {
      bf16* Cb = (bf16*)Cout;
      Cb[(size_t)m * ldc + nA] = (bf16)v0;
      Cb[(size_t)m * ldc + nB] = (bf16)v1;
    } else {
      float* Cf = (float*)Cout;
      Cf[(size_t)m * ldc + nA] = v0;
      Cf[(size_t)m * ldc + nB] = v1;
    }
  }
}

// ---------------------------------------------------------------------------
// Conversion / packing
// ---------------------------------------------------------------------------
__global__ void esbn_cvt_pad_bf16(const float* __restrict__ src, bf16* __restrict__ dst,
                                  int Ks, int Ns, int Kd, int Nd)
{
  int idx = blockIdx.x * blockDim.x + threadIdx.x;
  if (idx >= Kd * Nd) return;
  const int k = idx / Nd, n = idx % Nd;
  const float v = (k < Ks && n < Ns) ? src[(size_t)k * Ns + n] : 0.f;
  dst[idx] = (bf16)v;
}

__global__ void esbn_build_head_w(const float* __restrict__ keyw, const float* __restrict__ gw,
                                  const float* __restrict__ yw, bf16* __restrict__ dst)
{
  int idx = blockIdx.x * blockDim.x + threadIdx.x;   // HID * HEADN
  if (idx >= HID * HEADN) return;
  const int k = idx / HEADN, n = idx % HEADN;
  float v = 0.f;
  if (n < KEY)          v = keyw[(size_t)k * KEY + n];
  else if (n == KEY)    v = gw[k];
  else if (n < KEY + 5) v = yw[(size_t)k * 4 + (n - KEY - 1)];
  dst[idx] = (bf16)v;
}

__global__ void esbn_build_head_b(const float* __restrict__ keyb, const float* __restrict__ gb,
                                  const float* __restrict__ yb, float* __restrict__ dst)
{
  int n = blockIdx.x * blockDim.x + threadIdx.x;
  if (n >= HEADN) return;
  float v = 0.f;
  if (n < KEY)          v = keyb[n];
  else if (n == KEY)    v = gb[0];
  else if (n < KEY + 5) v = yb[n - KEY - 1];
  dst[n] = v;
}

__global__ void esbn_zero_u32(unsigned* __restrict__ p, size_t n)
{
  size_t i = (size_t)blockIdx.x * blockDim.x + threadIdx.x;
  const size_t stride = (size_t)gridDim.x * blockDim.x;
  for (; i < n; i += stride) p[i] = 0u;
}

// ---------------------------------------------------------------------------
// LSTM nonlinearity: h = sigmoid(o) * tanh(sigmoid(i) * tanh(c)); f gate dead.
// ---------------------------------------------------------------------------
__global__ void esbn_lstm_act(const float* __restrict__ gates, bf16* __restrict__ h_bf)
{
  int idx = blockIdx.x * blockDim.x + threadIdx.x;
  if (idx >= B_SZ * HID) return;
  const int b = idx >> 9, n = idx & (HID - 1);
  const float* g = gates + (size_t)b * GATESN;
  const float gi = g[n], gc = g[2 * HID + n], go = g[3 * HID + n];
  const float si = 1.f / (1.f + __expf(-gi));
  const float cn = si * tanhf(gc);
  const float so = 1.f / (1.f + __expf(-go));
  h_bf[idx] = (bf16)(so * tanhf(cn));
}

// ---------------------------------------------------------------------------
// One scan step per batch row: scores -> masked softmax -> key_r_new,
// slot writes, y_lin. 128 threads = 4 waves per block.
// ---------------------------------------------------------------------------
__global__ __launch_bounds__(128) void esbn_attention_step(
    const float* __restrict__ z_seq, const float* __restrict__ out2,
    float* __restrict__ M_k, float* __restrict__ M_v,
    bf16* __restrict__ key_r_bf, float* __restrict__ y_out,
    const float* __restrict__ conf_gain_p, const float* __restrict__ conf_bias_p, int t)
{
  __shared__ float s_z[Z_SZ];
  __shared__ float s_w[SLOTS];
  __shared__ float s_sc[SLOTS];
  __shared__ float s_ck[SLOTS];
  __shared__ float s_gate;

  const int b    = blockIdx.x;
  const int tid  = threadIdx.x;
  const int wave = tid >> 5;
  const int lane = tid & 31;

  const float zv = (t < T_SZ) ? z_seq[(size_t)b * T_SZ * Z_SZ + (size_t)t * Z_SZ + tid] : 0.f;
  s_z[tid] = zv;
  __syncthreads();

  const float* Mv_b = M_v + (size_t)b * SLOTS * Z_SZ;
  for (int s = wave; s < SLOTS; s += 4) {
    const float* row = Mv_b + (size_t)s * Z_SZ;
    float p = 0.f;
    #pragma unroll
    for (int j = 0; j < Z_SZ; j += 32) p += s_z[lane + j] * row[lane + j];
    #pragma unroll
    for (int off = 16; off; off >>= 1) p += __shfl_xor(p, off, 32);
    if (lane == 0) s_sc[s] = p;
  }
  __syncthreads();

  if (tid == 0) {
    const float cg = conf_gain_p[0], cb = conf_bias_p[0];
    float m = -3.4e38f;
    for (int s = 0; s < SLOTS; ++s) {
      const float ms = (s < t) ? s_sc[s] : NEGF;
      s_w[s] = ms;
      m = fmaxf(m, ms);
    }
    float sum = 0.f;
    for (int s = 0; s < SLOTS; ++s) { const float e = __expf(s_w[s] - m); s_w[s] = e; sum += e; }
    const float inv = 1.f / sum;
    for (int s = 0; s < SLOTS; ++s) {
      s_w[s] *= inv;
      s_ck[s] = 1.f / (1.f + __expf(-(s_sc[s] * cg + cb)));
    }
    s_gate = 1.f / (1.f + __expf(-out2[(size_t)b * HEADN + KEY]));
  }
  __syncthreads();

  const float* Mk_b = M_k + (size_t)b * SLOTS * KEY;
  const float gate = s_gate;
  for (int k = tid; k < KRP; k += 128) {
    float v = 0.f;
    if (k < KEY + 1 && t != 0) {
      float acc = 0.f;
      if (k < KEY) { for (int s = 0; s < SLOTS; ++s) acc += s_w[s] * Mk_b[(size_t)s * KEY + k]; }
      else         { for (int s = 0; s < SLOTS; ++s) acc += s_w[s] * s_ck[s]; }
      v = gate * acc;
    }
    key_r_bf[(size_t)b * KRP + k] = (bf16)v;   // pad cols 257..287 stay 0
  }

  for (int n = tid; n < KEY; n += 128)
    M_k[(size_t)b * SLOTS * KEY + (size_t)t * KEY + n] = out2[(size_t)b * HEADN + n];
  M_v[(size_t)b * SLOTS * Z_SZ + (size_t)t * Z_SZ + tid] = s_z[tid];
  if (tid < 4) y_out[(size_t)b * 4 + tid] = out2[(size_t)b * HEADN + KEY + 1 + tid];
}

__global__ void esbn_argmax(const float* __restrict__ y, float* __restrict__ out_idx)
{
  int b = blockIdx.x * blockDim.x + threadIdx.x;
  if (b >= B_SZ) return;
  const float* r = y + (size_t)b * 4;
  int best = 0; float bv = r[0];
  #pragma unroll
  for (int j = 1; j < 4; ++j) if (r[j] > bv) { bv = r[j]; best = j; }
  out_idx[b] = (float)best;
}

// ---------------------------------------------------------------------------
// Host side
// ---------------------------------------------------------------------------
static inline size_t esbn_align(size_t x) { return (x + 255) & ~(size_t)255; }

extern "C" void kernel_launch(void* const* d_in, const int* in_sizes, int n_in,
                              void* d_out, int out_size, void* d_ws, size_t ws_size,
                              hipStream_t stream)
{
  (void)in_sizes; (void)n_in; (void)out_size; (void)ws_size;

  const float* x_seq   = (const float*)d_in[0];
  const float* enc_w1  = (const float*)d_in[1];
  const float* enc_b1  = (const float*)d_in[2];
  const float* enc_w2  = (const float*)d_in[3];
  const float* enc_b2  = (const float*)d_in[4];
  const float* lstm_Wx = (const float*)d_in[5];
  const float* lstm_b  = (const float*)d_in[6];
  const float* keyw_W  = (const float*)d_in[7];
  const float* keyw_b  = (const float*)d_in[8];
  const float* g_W     = (const float*)d_in[9];
  const float* g_b     = (const float*)d_in[10];
  const float* y_W     = (const float*)d_in[11];
  const float* y_b     = (const float*)d_in[12];
  const float* cgain   = (const float*)d_in[13];
  const float* cbias   = (const float*)d_in[14];

  char* ws = (char*)d_ws;
  size_t off = 0;
  auto take = [&](size_t bytes) { char* p = ws + off; off = esbn_align(off + bytes); return p; };

  bf16*  x_bf     = (bf16*) take((size_t)B_SZ * T_SZ * D_SZ * 2);
  bf16*  w1_bf    = (bf16*) take((size_t)D_SZ * HID * 2);
  bf16*  w2_bf    = (bf16*) take((size_t)HID * Z_SZ * 2);
  bf16*  wx_bf    = (bf16*) take((size_t)KRP * GATESN * 2);
  bf16*  wh_bf    = (bf16*) take((size_t)HID * HEADN * 2);
  float* bh       = (float*)take((size_t)HEADN * 4);
  bf16*  hid1_bf  = (bf16*) take((size_t)B_SZ * T_SZ * HID * 2);
  float* z_seq    = (float*)take((size_t)B_SZ * T_SZ * Z_SZ * 4);
  bf16*  key_r_bf = (bf16*) take((size_t)B_SZ * KRP * 2);
  float* gates    = (float*)take((size_t)B_SZ * GATESN * 4);
  bf16*  h_bf     = (bf16*) take((size_t)B_SZ * HID * 2);
  float* out2     = (float*)take((size_t)B_SZ * HEADN * 4);
  float* M_k      = (float*)take((size_t)B_SZ * SLOTS * KEY * 4);
  float* M_v      = (float*)take((size_t)B_SZ * SLOTS * Z_SZ * 4);

  float* y_lin = (float*)d_out;              // (B,4)
  float* y_idx = (float*)d_out + B_SZ * 4;   // (B,)

  // ---- packing (every call: deterministic) ----
  esbn_cvt_pad_bf16<<<(B_SZ * T_SZ * D_SZ + 255) / 256, 256, 0, stream>>>(
      x_seq, x_bf, B_SZ * T_SZ, D_SZ, B_SZ * T_SZ, D_SZ);
  esbn_cvt_pad_bf16<<<(D_SZ * HID + 255) / 256, 256, 0, stream>>>(enc_w1, w1_bf, D_SZ, HID, D_SZ, HID);
  esbn_cvt_pad_bf16<<<(HID * Z_SZ + 255) / 256, 256, 0, stream>>>(enc_w2, w2_bf, HID, Z_SZ, HID, Z_SZ);
  esbn_cvt_pad_bf16<<<(KRP * GATESN + 255) / 256, 256, 0, stream>>>(lstm_Wx, wx_bf, KEY + 1, GATESN, KRP, GATESN);
  esbn_build_head_w<<<(HID * HEADN + 255) / 256, 256, 0, stream>>>(keyw_W, g_W, y_W, wh_bf);
  esbn_build_head_b<<<(HEADN + 255) / 256, 256, 0, stream>>>(keyw_b, g_b, y_b, bh);

  // ---- state init ----
  esbn_zero_u32<<<512, 256, 0, stream>>>((unsigned*)key_r_bf, (size_t)B_SZ * KRP * 2 / 4);
  esbn_zero_u32<<<1024, 256, 0, stream>>>((unsigned*)M_k, (size_t)B_SZ * SLOTS * KEY);
  esbn_zero_u32<<<1024, 256, 0, stream>>>((unsigned*)M_v, (size_t)B_SZ * SLOTS * Z_SZ);

  // ---- encoder ----
  {
    dim3 g1((B_SZ * T_SZ) / 64, HID / 64);
    esbn_gemm<1, 1><<<g1, 256, 0, stream>>>(x_bf, w1_bf, enc_b1, hid1_bf,
                                            B_SZ * T_SZ, HID, D_SZ, D_SZ, HID, HID);
    dim3 g2((B_SZ * T_SZ) / 64, Z_SZ / 64);
    esbn_gemm<0, 0><<<g2, 256, 0, stream>>>(hid1_bf, w2_bf, enc_b2, z_seq,
                                            B_SZ * T_SZ, Z_SZ, HID, HID, Z_SZ, Z_SZ);
  }

  // ---- sequential scan: t = 0 .. T inclusive ----
  for (int t = 0; t <= T_SZ; ++t) {
    dim3 gl(B_SZ / 64, GATESN / 64);
    esbn_gemm<0, 0><<<gl, 256, 0, stream>>>(key_r_bf, wx_bf, lstm_b, gates,
                                            B_SZ, GATESN, KRP, KRP, GATESN, GATESN);
    esbn_lstm_act<<<(B_SZ * HID + 255) / 256, 256, 0, stream>>>(gates, h_bf);
    dim3 gh(B_SZ / 64, HEADN / 64);
    esbn_gemm<0, 0><<<gh, 256, 0, stream>>>(h_bf, wh_bf, bh, out2,
                                            B_SZ, HEADN, HID, HID, HEADN, HEADN);
    esbn_attention_step<<<B_SZ, 128, 0, stream>>>(z_seq, out2, M_k, M_v,
                                                  key_r_bf, y_lin, cgain, cbias, t);
  }

  esbn_argmax<<<(B_SZ + 255) / 256, 256, 0, stream>>>(y_lin, y_idx);
}
